// BitNetLinear_75857712382176
// MI455X (gfx1250) — compile-verified, matrix operand order checked by reading, not run on hardware
//
#include <hip/hip_runtime.h>
#include <hip/hip_bf16.h>
#include <stdint.h>

// BitNet ternary linear: out[M,N] = (x[M,K] @ W[N,K]^T) * scale[n]
// M = B*S = 8192, K = 4096, N = 11008
#define M_DIM 8192
#define K_DIM 4096
#define N_DIM 11008
#define KP    (K_DIM / 4)   // packed bytes per weight row (1024)

typedef __attribute__((ext_vector_type(16))) __bf16        v16bf;
typedef __attribute__((ext_vector_type(8)))  float         v8f;
typedef __attribute__((ext_vector_type(4)))  unsigned int  u32x4;

union FragA { v16bf v; u32x4 q[2]; };
union FragB { v16bf v; unsigned long long d[4]; };
union PackW { unsigned long long u64; unsigned int u32[2]; };

__global__ __launch_bounds__(256)
void bitnet_wmma_kernel(const __hip_bfloat16* __restrict__ X,
                        const unsigned char*  __restrict__ Wp,
                        const __hip_bfloat16* __restrict__ Sc,
                        __hip_bfloat16*       __restrict__ Out)
{
    // ---- 256-entry byte -> 4x bf16 ternary decode LUT in LDS (2 KB) ----
    // code 0 -> 0.0, 1 -> +1.0 (0x3F80), 2 -> -1.0 (0xBF80)
    __shared__ unsigned long long lut[256];
    {
        const unsigned t = threadIdx.x;      // exactly 256 threads
        unsigned long long e = 0ull;
        #pragma unroll
        for (int j = 0; j < 4; ++j) {
            const unsigned c  = (t >> (2 * j)) & 3u;
            const unsigned nz = (c | (c >> 1)) & 1u;                // c != 0
            const unsigned hv = (nz * 0x3F80u) | ((c & 2u) << 14);  // sign from bit1
            e |= ((unsigned long long)(hv & 0xFFFFu)) << (16 * j);
        }
        lut[t] = e;
    }
    __syncthreads();

    const int lane = threadIdx.x & 31;
    const int wave = threadIdx.x >> 5;   // 8 waves
    const int wm   = wave >> 2;          // 0..1 : M group within block
    const int wn   = wave & 3;           // 0..3 : N group within block
    const int half = lane >> 4;          // 0/1  : which 16-lane half
    const int l16  = lane & 15;

    const int m0 = blockIdx.x * 128 + wm * 64;   // wave computes a 64 x 64 tile
    const int n0 = blockIdx.y * 256 + wn * 64;

    v8f acc[4][4] = {};   // 16 tiles of 16x16 f32 = 128 VGPRs

    // A lane base: row (m0 + l16), starting at K element half*8.
    const __hip_bfloat16* xbase = X + (size_t)(m0 + l16) * K_DIM + half * 8;
    // B lane base: packed row (n0 + l16); rows for j>0 are +j*16*KP (imm offset).
    const unsigned char*  wbase = Wp + (size_t)(n0 + l16) * KP;

    // v_perm selector: gather packed-weight bytes {2h, 2h+1, 4+2h, 5+2h}
    // (byte idx 0-3 -> lo dword, 4-7 -> hi dword) into one dword.
    const unsigned bsel = (unsigned)(2 * half)
                        | ((unsigned)(2 * half + 1) << 8)
                        | ((unsigned)(2 * half + 4) << 16)
                        | ((unsigned)(2 * half + 5) << 24);

    for (int k0 = 0; k0 < K_DIM; k0 += 32) {
        // ---- load 4 A fragments (16x32 bf16 each) ----
        FragA a[4];
        #pragma unroll
        for (int i = 0; i < 4; ++i) {
            const __hip_bfloat16* xr = xbase + (size_t)i * 16 * K_DIM + k0;
            a[i].q[0] = *(const u32x4*)(xr);        // elems 0..7  : K = k0 + half*8 ..
            a[i].q[1] = *(const u32x4*)(xr + 16);   // elems 8..15 : K = k0 + 16 + half*8 ..
        }

        // ---- load + decode 4 B fragments (32x16 bf16, one column per lane) ----
        FragB b[4];
        #pragma unroll
        for (int j = 0; j < 4; ++j) {
            PackW p;
            p.u64 = *(const unsigned long long*)(wbase + (size_t)j * 16 * KP + (k0 >> 2));
            // gather the 4 bytes this half-lane needs into one dword
            const unsigned q = __builtin_amdgcn_perm(p.u32[1], p.u32[0], bsel);
            b[j].d[0] = lut[ q         & 0xFFu];  // K local half*8     .. +3
            b[j].d[1] = lut[(q >>  8)  & 0xFFu];  // K local half*8 + 4 .. +7
            b[j].d[2] = lut[(q >> 16)  & 0xFFu];  // K local 16+half*8  .. +3
            b[j].d[3] = lut[ q >> 24         ];   // K local 16+half*8+4.. +7
        }

        // ---- 16 WMMAs: 64x64 tile advances by K=32 ----
        #pragma unroll
        for (int i = 0; i < 4; ++i) {
            #pragma unroll
            for (int j = 0; j < 4; ++j) {
                acc[i][j] = __builtin_amdgcn_wmma_f32_16x16x32_bf16(
                    false, a[i].v, false, b[j].v,
                    (short)0, acc[i][j], false, false);
            }
        }
    }

    // ---- epilogue: per-column scale, f32 -> bf16 ----
    // C/D layout: lane owns column n = n0 + j*16 + l16; VGPR r holds row
    // m = m0 + i*16 + half*8 + r.
    #pragma unroll
    for (int j = 0; j < 4; ++j) {
        const int n = n0 + j * 16 + l16;
        const float s = __bfloat162float(Sc[n]);
        #pragma unroll
        for (int i = 0; i < 4; ++i) {
            __hip_bfloat16* orow =
                Out + (size_t)(m0 + i * 16 + half * 8) * N_DIM + n;
            #pragma unroll
            for (int r = 0; r < 8; ++r) {
                orow[(size_t)r * N_DIM] = __float2bfloat16(acc[i][j][r] * s);
            }
        }
    }
}

extern "C" void kernel_launch(void* const* d_in, const int* in_sizes, int n_in,
                              void* d_out, int out_size, void* d_ws, size_t ws_size,
                              hipStream_t stream) {
    const __hip_bfloat16* x   = (const __hip_bfloat16*)d_in[0];
    const unsigned char*  wp  = (const unsigned char*)d_in[1];
    const __hip_bfloat16* sc  = (const __hip_bfloat16*)d_in[2];
    __hip_bfloat16*       out = (__hip_bfloat16*)d_out;

    dim3 grid(M_DIM / 128, N_DIM / 256);   // 64 x 43, exact tiling
    dim3 block(256, 1, 1);                 // 8 waves
    bitnet_wmma_kernel<<<grid, block, 0, stream>>>(x, wp, sc, out);
}
